// NDCN_odefunc_52913997087306
// MI455X (gfx1250) — matrix operand order repaired and checked by reading, not compiled, not run on hardware
//
#include <hip/hip_runtime.h>
#include <hip/hip_bf16.h>
#include <math.h>

// GCNConv + tanh for MI455X (gfx1250, wave32, WMMA).
// h = x@W via v_wmma_f32_16x16x32_bf16 with 3xBF16 split (fp32-faithful),
// edge scatter via f32 atomics (L2-resident on 192MB L2), tanh finalize.

typedef __attribute__((ext_vector_type(16))) __bf16 v16bf;
typedef __attribute__((ext_vector_type(8)))  float  v8f;

#define D 128

// ---------------- kernel 1: zero accumulator + degree ----------------
__global__ void gcn_zero_kernel(float* __restrict__ acc, int* __restrict__ deg,
                                int nAcc, int nDeg) {
  int i = blockIdx.x * blockDim.x + threadIdx.x;
  if (i < nAcc) acc[i] = 0.0f;
  if (i < nDeg) deg[i] = 0;
}

// ---------------- kernel 2: in-degree count (targets) ----------------
__global__ void gcn_deg_kernel(const long long* __restrict__ ei,
                               int* __restrict__ deg, int E) {
  int e = blockIdx.x * blockDim.x + threadIdx.x;
  if (e < E) {
    int c = (int)ei[(size_t)E + e];  // edge_index[1][e] = target
    atomicAdd(&deg[c], 1);
  }
}

// ---------------- kernel 3: dinv = rsqrt(deg + 1 self-loop) ----------
__global__ void gcn_dinv_kernel(const int* __restrict__ deg,
                                float* __restrict__ dinv, int n) {
  int i = blockIdx.x * blockDim.x + threadIdx.x;
  if (i < n) dinv[i] = rsqrtf((float)deg[i] + 1.0f);
}

// ---------------- kernel 4: h = x @ W via WMMA (3xBF16 split) --------
// Block = 256 threads = 8 waves. Each wave computes one 16-row tile x all
// 128 output columns (8 accumulators of 16x16 f32).
// W is pre-split hi/lo bf16 and swizzled into LDS in B-fragment order so a
// lane's 16 operand halves are 32 contiguous bytes.
__global__ void __launch_bounds__(256)
gcn_gemm_wmma_kernel(const float* __restrict__ x, const float* __restrict__ W,
                     float* __restrict__ h, int nTiles) {
  // [kc(4)][nt(8)][lane(32)][j(16)] bf16 -> 32KB each, 64KB LDS total
  __shared__ __align__(32) __bf16 sWhi[4 * 8 * 32 * 16];
  __shared__ __align__(32) __bf16 sWlo[4 * 8 * 32 * 16];

  // Cooperative fill: B-matrix ISA layout (16-bit B, 32x16):
  //   lanes 0-15: j=0..7 -> K 0..7, j=8..15 -> K 16..23;
  //   lanes 16-31: j=0..7 -> K 8..15, j=8..15 -> K 24..31; N = lane & 15.
  for (int idx = threadIdx.x; idx < 16384; idx += 256) {
    int kc = idx >> 12;        // k-chunk (32 K per chunk)
    int r  = idx & 4095;
    int nt = r >> 9;           // 16-col tile
    int r2 = r & 511;
    int l  = r2 >> 4;          // lane
    int j  = r2 & 15;          // half index within lane
    int K    = kc * 32 + (j & 7) + ((j >= 8) ? 16 : 0) + ((l >= 16) ? 8 : 0);
    int Ncol = nt * 16 + (l & 15);
    float w = W[K * D + Ncol];
    __bf16 hi = (__bf16)w;
    sWhi[idx] = hi;
    sWlo[idx] = (__bf16)(w - (float)hi);   // residual
  }
  __syncthreads();

  int wave = threadIdx.x >> 5;
  int lane = threadIdx.x & 31;
  int tile = blockIdx.x * 8 + wave;
  if (tile >= nTiles) return;   // wave-uniform branch: EXEC stays all-ones
  int rowBase = tile * 16;

  union F8  { float  f[8];  v8f   v; };
  union B16 { __bf16 b[16]; v16bf v; };

  F8 acc[8];
  #pragma unroll
  for (int nt = 0; nt < 8; ++nt)
    #pragma unroll
    for (int k = 0; k < 8; ++k) acc[nt].f[k] = 0.0f;

  // A-matrix ISA layout (16-bit A, 16x32): lane m=lane&15;
  // lanes 0-15: j=0..7 -> K 0..7, j=8..15 -> K 16..23
  // lanes 16-31: j=0..7 -> K 8..15, j=8..15 -> K 24..31
  int mrow  = rowBase + (lane & 15);
  int kbase = (lane >= 16) ? 8 : 0;

  for (int kc = 0; kc < 4; ++kc) {
    B16 ahi, alo;
    const float* xr = x + (size_t)mrow * D + kc * 32 + kbase;
    #pragma unroll
    for (int j = 0; j < 8; ++j) {       // K = kbase + 0..7
      float f = xr[j];
      __bf16 hi = (__bf16)f;
      ahi.b[j] = hi;
      alo.b[j] = (__bf16)(f - (float)hi);
    }
    #pragma unroll
    for (int j = 0; j < 8; ++j) {       // K = kbase + 16..23
      float f = xr[16 + j];
      __bf16 hi = (__bf16)f;
      ahi.b[8 + j] = hi;
      alo.b[8 + j] = (__bf16)(f - (float)hi);
    }

    #pragma unroll
    for (int nt = 0; nt < 8; ++nt) {
      const v16bf* bh = (const v16bf*)&sWhi[(size_t)(kc * 8 + nt) * 512 + lane * 16];
      const v16bf* bl = (const v16bf*)&sWlo[(size_t)(kc * 8 + nt) * 512 + lane * 16];
      v16bf bhv = *bh;
      v16bf blv = *bl;
      // D = A*B + C ; accumulate hi*hi + hi*lo + lo*hi in f32
      acc[nt].v = __builtin_amdgcn_wmma_f32_16x16x32_bf16(
          false, ahi.v, false, bhv, (short)0, acc[nt].v, false, false);
      acc[nt].v = __builtin_amdgcn_wmma_f32_16x16x32_bf16(
          false, ahi.v, false, blv, (short)0, acc[nt].v, false, false);
      acc[nt].v = __builtin_amdgcn_wmma_f32_16x16x32_bf16(
          false, alo.v, false, bhv, (short)0, acc[nt].v, false, false);
    }
  }

  // C/D f32 16x16 layout: VGPR v -> M = v + (lane>=16 ? 8 : 0), N = lane&15
  int mbase = rowBase + ((lane >= 16) ? 8 : 0);
  #pragma unroll
  for (int nt = 0; nt < 8; ++nt) {
    int ncol = nt * 16 + (lane & 15);
    #pragma unroll
    for (int v = 0; v < 8; ++v)
      h[(size_t)(mbase + v) * D + ncol] = acc[nt].f[v];
  }
}

// ---------------- kernel 5: edge scatter with f32 atomics ------------
// One wave per edge; lane covers 4 columns (32*4 = 128).
__global__ void __launch_bounds__(256)
gcn_scatter_kernel(const long long* __restrict__ ei,
                   const float* __restrict__ dinv,
                   const float* __restrict__ h,
                   float* __restrict__ acc, int E) {
  int wave = threadIdx.x >> 5;
  int lane = threadIdx.x & 31;
  long long e = (long long)blockIdx.x * 8 + wave;
  if (e >= E) return;
  long long r = ei[e];               // source
  long long c = ei[(size_t)E + e];   // target
  float nrm = dinv[r] * dinv[c];
  const float4* hr = (const float4*)(h + (size_t)r * D);
  float4 v = hr[lane];
  float* op = acc + (size_t)c * D + lane * 4;
  unsafeAtomicAdd(op + 0, v.x * nrm);
  unsafeAtomicAdd(op + 1, v.y * nrm);
  unsafeAtomicAdd(op + 2, v.z * nrm);
  unsafeAtomicAdd(op + 3, v.w * nrm);
}

// ---------------- kernel 6: self-loop + bias + tanh ------------------
__global__ void gcn_final_kernel(const float* __restrict__ acc,
                                 const float* __restrict__ h,
                                 const float* __restrict__ dinv,
                                 const float* __restrict__ b,
                                 float* __restrict__ out, int n) {
  int i = blockIdx.x * blockDim.x + threadIdx.x;
  if (i >= n) return;
  int node = i >> 7;   // /128
  int col  = i & 127;
  float di = dinv[node];
  out[i] = tanhf(acc[i] + h[i] * di * di + b[col]);
}

extern "C" void kernel_launch(void* const* d_in, const int* in_sizes, int n_in,
                              void* d_out, int out_size, void* d_ws, size_t ws_size,
                              hipStream_t stream) {
  // inputs: 0=t(unused), 1=x[N,128] f32, 2=edge_index[2,E] i64, 3=W[128,128] f32, 4=b[128] f32
  const float*     x  = (const float*)d_in[1];
  const long long* ei = (const long long*)d_in[2];
  const float*     W  = (const float*)d_in[3];
  const float*     b  = (const float*)d_in[4];
  float* out = (float*)d_out;

  int N = in_sizes[1] / D;   // 50000
  int E = in_sizes[2] / 2;   // 800000

  // workspace layout
  char* ws = (char*)d_ws;
  size_t o_deg  = 0;
  size_t o_dinv = (o_deg  + (size_t)N * sizeof(int)       + 255) & ~(size_t)255;
  size_t o_h    = (o_dinv + (size_t)N * sizeof(float)     + 255) & ~(size_t)255;
  size_t o_acc  = (o_h    + (size_t)N * D * sizeof(float) + 255) & ~(size_t)255;
  int*   deg  = (int*)(ws + o_deg);
  float* dinv = (float*)(ws + o_dinv);
  float* h    = (float*)(ws + o_h);
  float* acc  = (float*)(ws + o_acc);

  int nAcc = N * D;

  // 1. zero acc + deg
  {
    int n = (nAcc > N) ? nAcc : N;
    gcn_zero_kernel<<<(n + 255) / 256, 256, 0, stream>>>(acc, deg, nAcc, N);
  }
  // 2. degree count
  gcn_deg_kernel<<<(E + 255) / 256, 256, 0, stream>>>(ei, deg, E);
  // 3. dinv
  gcn_dinv_kernel<<<(N + 255) / 256, 256, 0, stream>>>(deg, dinv, N);
  // 4. GEMM h = x @ W (WMMA)
  {
    int nTiles = (N + 15) / 16;            // 3125 (exact: N % 16 == 0)
    int blocks = (nTiles + 7) / 8;
    gcn_gemm_wmma_kernel<<<blocks, 256, 0, stream>>>(x, W, h, nTiles);
  }
  // 5. edge scatter
  gcn_scatter_kernel<<<(E + 7) / 8, 256, 0, stream>>>(ei, dinv, h, acc, E);
  // 6. finalize
  gcn_final_kernel<<<(nAcc + 255) / 256, 256, 0, stream>>>(acc, h, dinv, b, out, nAcc);
}
